// Rudy_79362405696090
// MI455X (gfx1250) — matrix operand order repaired and checked by reading, not compiled, not run on hardware
//
#include <hip/hip_runtime.h>

typedef __attribute__((ext_vector_type(16))) _Float16 v16h;
typedef __attribute__((ext_vector_type(8)))  float    v8f;

#define NBX 512
#define NBY 512
#define BSX (1000.0f/512.0f)
#define BSY (1000.0f/512.0f)
#define NHT 100.0f
#define NVT 100.0f
#define MAXR 2.0f
#define MINR 0.5f

// ---------------------------------------------------------------------------
// Kernel 1: per-net bounding boxes + RUDY weights (SoA, stride ntot).
// P[0..5][ntot] = xmin, xmax, ymin, ymax, wh=w/(ymax-ymin), wv=w/(xmax-xmin).
// Indices [num_nets, ntot) get degenerate params (xmin>xmax, zero weights) so
// the GEMM can load parameters unconditionally for every K chunk.
// ---------------------------------------------------------------------------
__global__ void rudy_bbox(const float* __restrict__ pin_pos,
                          const int*   __restrict__ netpin_start,
                          const int*   __restrict__ flat_netpin,
                          const float* __restrict__ net_w,
                          float* __restrict__ P,
                          int num_nets, int ntot, int num_pins)
{
    int n = blockIdx.x * blockDim.x + threadIdx.x;
    if (n >= ntot) return;
    if (n >= num_nets) {                    // pad region: contributes nothing
        P[0 * ntot + n] = 1.0f;             // xmin > xmax  -> ox = 0
        P[1 * ntot + n] = 0.0f;
        P[2 * ntot + n] = 1.0f;             // ymin > ymax  -> oy = 0
        P[3 * ntot + n] = 0.0f;
        P[4 * ntot + n] = 0.0f;
        P[5 * ntot + n] = 0.0f;
        return;
    }
    int s = netpin_start[n];
    int e = netpin_start[n + 1];
    float xmn =  3.4e38f, xmx = -3.4e38f;
    float ymn =  3.4e38f, ymx = -3.4e38f;
    for (int i = s; i < e; ++i) {
        int p = flat_netpin[i];
        float x = pin_pos[p];
        float y = pin_pos[num_pins + p];
        xmn = fminf(xmn, x); xmx = fmaxf(xmx, x);
        ymn = fminf(ymn, y); ymx = fmaxf(ymx, y);
    }
    float w = net_w[n];
    P[0 * ntot + n] = xmn;
    P[1 * ntot + n] = xmx;
    P[2 * ntot + n] = ymn;
    P[3 * ntot + n] = ymx;
    P[4 * ntot + n] = w / (ymx - ymn);
    P[5 * ntot + n] = w / (xmx - xmn);
}

// Load a 16-half WMMA fragment from two 16-byte aligned LDS locations.
__device__ __forceinline__ v16h ldfrag(const _Float16* p0, const _Float16* p1)
{
    union U { v16h v; float4 f[2]; } u;
    u.f[0] = *reinterpret_cast<const float4*>(p0);
    u.f[1] = *reinterpret_cast<const float4*>(p1);
    return u.v;
}

// ---------------------------------------------------------------------------
// Kernel 2: H = (ox*wh)^T * oy, V = (ox*wv)^T * oy as WMMA GEMMs, K = nets.
// Block = 256 threads (8 wave32). 128x128 output tile per block (both maps),
// K-chunked over blockIdx.z with f32 atomic accumulation. A carries the
// per-net weights so a single unweighted B tile is shared by both maps.
// Tiles are generated on the fly in LDS from 6 floats/net (no HBM A/B).
// Each thread owns net k = lane for the whole step; params are loaded
// unconditionally (P is padded) into registers before the barrier.
// ---------------------------------------------------------------------------
__launch_bounds__(256)
__global__ void rudy_gemm(const float* __restrict__ P,
                          float* __restrict__ hacc,
                          float* __restrict__ vacc,
                          int ntot, int nets_per_chunk)
{
    // A tiles: [128 m][32 k] halfs, rows padded to 40 to spread LDS banks.
    // B tile : [32 k][128 n] halfs, rows padded to 136.
    __shared__ __align__(16) _Float16 Ah[128 * 40];
    __shared__ __align__(16) _Float16 Av[128 * 40];
    __shared__ __align__(16) _Float16 Bs[32 * 136];

    const int tid  = threadIdx.x;
    const int lane = tid & 31;
    const int grow = tid >> 5;          // 0..7 : generator row group
    const int wave = tid >> 5;
    const int wm   = (wave >> 1) << 5;  // 0,32,64,96 : wave row strip (x)
    const int wn   = (wave &  1) << 6;  // 0,64       : wave col half  (y)

    const int bx0 = blockIdx.x * 128;
    const int by0 = blockIdx.y * 128;

    const int k0   = blockIdx.z * nets_per_chunk;
    const int kend = k0 + nets_per_chunk;     // always <= ntot (host-padded)

    v8f aH[2][4] = {};
    v8f aV[2][4] = {};

    for (int kk = k0; kk < kend; kk += 32) {
        // This thread's net for the whole step (k == lane in the tile).
        const int nz = kk + lane;
        const float xmn = P[0 * ntot + nz];
        const float xmx = P[1 * ntot + nz];
        const float ymn = P[2 * ntot + nz];
        const float ymx = P[3 * ntot + nz];
        const float whk = P[4 * ntot + nz];
        const float wvk = P[5 * ntot + nz];

        __syncthreads();                 // previous iter's fragment reads done

        // A tiles: 128 x-bins for this thread's net (weighted twice).
#pragma unroll
        for (int i = 0; i < 16; ++i) {
            int m = grow + (i << 3);
            float el = BSX * (float)(bx0 + m);
            float ox = fmaxf(fminf(xmx, el + BSX) - fmaxf(xmn, el), 0.0f);
            Ah[m * 40 + lane] = (_Float16)(ox * whk);
            Av[m * 40 + lane] = (_Float16)(ox * wvk);
        }
        // B tile: 128 y-bins for this thread's net (unweighted).
#pragma unroll
        for (int i = 0; i < 16; ++i) {
            int n = grow + (i << 3);
            float el = BSY * (float)(by0 + n);
            float oy = fmaxf(fminf(ymx, el + BSY) - fmaxf(ymn, el), 0.0f);
            Bs[lane * 136 + n] = (_Float16)oy;
        }
        __syncthreads();

        // A fragments (16-bit A 16x32 layout): lane = M (mod 16), element e ->
        // K = (e>>3)*16 + (lane>>4)*8 + (e&7).
        const int am  = wm + (lane & 15);
        const int akb = (lane >> 4) * 8;
        const _Float16* ph0 = &Ah[am * 40 + akb];
        const _Float16* ph1 = &Ah[(am + 16) * 40 + akb];
        const _Float16* pv0 = &Av[am * 40 + akb];
        const _Float16* pv1 = &Av[(am + 16) * 40 + akb];
        v16h fh[2], fv[2];
        fh[0] = ldfrag(ph0, ph0 + 16);
        fh[1] = ldfrag(ph1, ph1 + 16);
        fv[0] = ldfrag(pv0, pv0 + 16);
        fv[1] = ldfrag(pv1, pv1 + 16);

        // B fragments (16-bit B 32x16 layout): lane = N (mod 16), element e ->
        // K = (e>>3)*16 + (lane>>4)*8 + (e&7).
        v16h fb[4];
#pragma unroll
        for (int t = 0; t < 4; ++t) {
            const int bn = wn + 16 * t + (lane & 15);
#pragma unroll
            for (int e = 0; e < 16; ++e)
                fb[t][e] = Bs[((lane >> 4) * 8 + (e & 7) + (e >> 3) * 16) * 136 + bn];
        }

#pragma unroll
        for (int s = 0; s < 2; ++s) {
#pragma unroll
            for (int t = 0; t < 4; ++t) {
                aH[s][t] = __builtin_amdgcn_wmma_f32_16x16x32_f16(
                    false, fh[s], false, fb[t], (short)0, aH[s][t], false, false);
                aV[s][t] = __builtin_amdgcn_wmma_f32_16x16x32_f16(
                    false, fv[s], false, fb[t], (short)0, aV[s][t], false, false);
            }
        }
    }

    // C/D 16x16 f32 layout: VGPR r, lanes 0-15: M=r, N=lane; lanes 16-31: M=r+8.
    const int mloc = (lane >> 4) << 3;
    const int nloc = lane & 15;
#pragma unroll
    for (int s = 0; s < 2; ++s) {
#pragma unroll
        for (int t = 0; t < 4; ++t) {
#pragma unroll
            for (int r = 0; r < 8; ++r) {
                int x = bx0 + wm + 16 * s + mloc + r;
                int y = by0 + wn + 16 * t + nloc;
                unsafeAtomicAdd(&hacc[x * NBY + y], aH[s][t][r]);
                unsafeAtomicAdd(&vacc[x * NBY + y], aV[s][t][r]);
            }
        }
    }
}

// ---------------------------------------------------------------------------
// Kernel 3: scale, |.|, max, square, clip -> d_out[x*512+y]
// ---------------------------------------------------------------------------
__global__ void rudy_final(const float* __restrict__ hacc,
                           const float* __restrict__ vacc,
                           float* __restrict__ out)
{
    int i = blockIdx.x * blockDim.x + threadIdx.x;
    if (i >= NBX * NBY) return;
    float h = hacc[i] * (1.0f / (BSX * NHT));
    float v = vacc[i] * (1.0f / (BSY * NVT));
    float r = fmaxf(fabsf(h), fabsf(v));
    out[i] = fminf(fmaxf(r * r, MINR), MAXR);
}

extern "C" void kernel_launch(void* const* d_in, const int* in_sizes, int n_in,
                              void* d_out, int out_size, void* d_ws, size_t ws_size,
                              hipStream_t stream)
{
    const float* pin_pos      = (const float*)d_in[0];
    const int*   netpin_start = (const int*)  d_in[1];
    const int*   flat_netpin  = (const int*)  d_in[2];
    const float* net_w        = (const float*)d_in[3];
    const int num_pins = in_sizes[2];
    const int num_nets = in_sizes[3];

    const int KCHUNKS = 64;                        // (4,4,64) = 1024 blocks
    int npc = (num_nets + KCHUNKS - 1) / KCHUNKS;
    npc = (npc + 31) & ~31;                        // multiple of WMMA K
    const int ntot = npc * KCHUNKS;                // padded net count

    float* P    = (float*)d_ws;                    // 6 * ntot
    float* hacc = P + (size_t)6 * ntot;            // 512*512
    float* vacc = hacc + (size_t)NBX * NBY;        // 512*512

    hipMemsetAsync(hacc, 0, (size_t)2 * NBX * NBY * sizeof(float), stream);

    int nb = (ntot + 255) / 256;
    rudy_bbox<<<nb, 256, 0, stream>>>(pin_pos, netpin_start, flat_netpin, net_w,
                                      P, num_nets, ntot, num_pins);

    dim3 grid(NBX / 128, NBY / 128, KCHUNKS);
    rudy_gemm<<<grid, 256, 0, stream>>>(P, hacc, vacc, ntot, npc);

    rudy_final<<<(NBX * NBY + 255) / 256, 256, 0, stream>>>(hacc, vacc, (float*)d_out);
}
